// Transformer_64699387347290
// MI455X (gfx1250) — matrix operand order repaired and checked by reading, not compiled, not run on hardware
//
#include <hip/hip_runtime.h>

// ---------------------------------------------------------------------------
// Problem constants (from the reference)
// ---------------------------------------------------------------------------
#define BB 4
#define TT 2048
#define HH_ 16
#define DKK 64
#define FF 1024
#define CKK 64
#define NLL 16

typedef __attribute__((ext_vector_type(16))) _Float16 v16h;
typedef __attribute__((ext_vector_type(8)))  _Float16 v8h;
typedef __attribute__((ext_vector_type(8)))  float    v8f;

// CDNA5 async global->LDS DMA (ASYNCcnt-tracked), guarded so the file always
// compiles even if this toolchain doesn't declare the builtins.
#if defined(__has_builtin)
#if __has_builtin(__builtin_amdgcn_global_load_async_to_lds_b128) && \
    __has_builtin(__builtin_amdgcn_s_wait_asynccnt)
#define HAVE_ASYNC_LDS 1
#endif
#endif
#ifndef HAVE_ASYNC_LDS
#define HAVE_ASYNC_LDS 0
#endif

#if HAVE_ASYNC_LDS
typedef int v4i_vs __attribute__((vector_size(4 * sizeof(int))));
typedef __attribute__((address_space(1))) v4i_vs* as1_v4i;  // global
typedef __attribute__((address_space(3))) v4i_vs* as3_v4i;  // LDS
#endif

// Copy 16 f16 (32B) global -> LDS.
static __device__ inline void cp16_g2s(_Float16* dst, const _Float16* src) {
#if HAVE_ASYNC_LDS
  __builtin_amdgcn_global_load_async_to_lds_b128((as1_v4i)src, (as3_v4i)dst, 0, 0);
  __builtin_amdgcn_global_load_async_to_lds_b128((as1_v4i)(src + 8),
                                                 (as3_v4i)(dst + 8), 0, 0);
#else
  ((v8h*)dst)[0] = ((const v8h*)src)[0];
  ((v8h*)dst)[1] = ((const v8h*)src)[1];
#endif
}

static __device__ inline void async_fence() {
#if HAVE_ASYNC_LDS
  __builtin_amdgcn_s_wait_asynccnt(0);
#endif
}

// B fragment: 16 contiguous f16 (order used here: lane half h holds K = h*16 + i)
static __device__ inline v16h load_b_frag(const _Float16* p) {
  v16h r;
#pragma unroll
  for (int i = 0; i < 16; ++i) r[i] = p[i];
  return r;
}

// A fragment: two contiguous runs of 8 f16
// (order used here: lane half h holds K = {h*16+0..7} in elems 0..7, {h*16+8..15} in 8..15)
static __device__ inline v16h load_a_frag(const _Float16* p0, const _Float16* p1) {
  v16h r;
#pragma unroll
  for (int i = 0; i < 8; ++i) { r[i] = p0[i]; r[8 + i] = p1[i]; }
  return r;
}

static __device__ inline v8f wmma16(v16h a, v16h b, v8f c) {
  return __builtin_amdgcn_wmma_f32_16x16x32_f16(false, a, false, b, (short)0, c,
                                                false, false);
}

static __device__ inline v8f zero8() {
  v8f z = {0.f, 0.f, 0.f, 0.f, 0.f, 0.f, 0.f, 0.f};
  return z;
}

// ---------------------------------------------------------------------------
// Projection GEMM:  Out[m,n] = sum_k A[m,k] * W[n,k] + bias[n]
// A: (M,K) row-major (f32 or f16), W: (N,K) row-major f32.
// Block = 256 threads = 8 waves, 64x64 output tile, k-step 32.
// ---------------------------------------------------------------------------
template <typename AT, bool OUT16>
__global__ __launch_bounds__(256) void proj_gemm(
    const AT* __restrict__ A, const float* __restrict__ W,
    const float* __restrict__ bias, void* __restrict__ Out, int M, int K, int N) {
  __shared__ _Float16 As[64][40];  // 64 x 32, padded
  __shared__ _Float16 Bs[64][40];

  const int tid  = threadIdx.x;
  const int lane = tid & 31;
  const int wid  = tid >> 5;
  const int wm   = wid >> 1;
  const int wn   = wid & 1;
  const int hh   = lane >> 4;
  const int ln   = lane & 15;
  const int bm   = blockIdx.y * 64;
  const int bn   = blockIdx.x * 64;

  v8f c0 = zero8(), c1 = zero8();

  const int ldr = tid >> 2;       // 0..63
  const int ldc = (tid & 3) * 8;  // 0,8,16,24

  for (int k0 = 0; k0 < K; k0 += 32) {
    __syncthreads();
    {
      const AT* srcA = A + (size_t)(bm + ldr) * K + k0 + ldc;
      const float* srcW = W + (size_t)(bn + ldr) * K + k0 + ldc;
#pragma unroll
      for (int i = 0; i < 8; ++i) As[ldr][ldc + i] = (_Float16)(float)srcA[i];
#pragma unroll
      for (int i = 0; i < 8; ++i) Bs[ldr][ldc + i] = (_Float16)srcW[i];
    }
    __syncthreads();

    const int arow = wm * 16 + ln;
    v16h a  = load_a_frag(&As[arow][hh * 16], &As[arow][hh * 16 + 8]);
    v16h b0 = load_b_frag(&Bs[wn * 32 + ln][hh * 16]);
    v16h b1 = load_b_frag(&Bs[wn * 32 + 16 + ln][hh * 16]);
    c0 = wmma16(a, b0, c0);
    c1 = wmma16(a, b1, c1);
  }

  const int col0 = bn + wn * 32 + ln;
  const int col1 = col0 + 16;
  const float bb0 = bias ? bias[col0] : 0.f;
  const float bb1 = bias ? bias[col1] : 0.f;
#pragma unroll
  for (int r = 0; r < 8; ++r) {
    const int row = bm + wm * 16 + hh * 8 + r;
    const float v0 = c0[r] + bb0;
    const float v1 = c1[r] + bb1;
    if (OUT16) {
      _Float16* o = (_Float16*)Out;
      o[(size_t)row * N + col0] = (_Float16)v0;
      o[(size_t)row * N + col1] = (_Float16)v1;
    } else {
      float* o = (float*)Out;
      o[(size_t)row * N + col0] = v0;
      o[(size_t)row * N + col1] = v1;
    }
  }
}

// ---------------------------------------------------------------------------
// Fused banded flash attention per (query-chunk, head, batch).
// Block = 128 threads (4 waves); wave w owns query rows [w*16, w*16+16).
// S = (q+u)K^T + (q+v)P^T, online softmax, O += P_exp @ V.
// K and P chunks are staged into LDS via async global->LDS DMA; the manual
// transposed V staging overlaps with the DMA.
// ---------------------------------------------------------------------------
__global__ __launch_bounds__(128) void attn_fused(
    const _Float16* __restrict__ Q, const _Float16* __restrict__ Kh,
    const _Float16* __restrict__ Vh, const _Float16* __restrict__ Ph,
    const float* __restrict__ bias_u, const float* __restrict__ bias_v,
    _Float16* __restrict__ X) {
  __shared__ _Float16 QU[64][72];
  __shared__ _Float16 QV[64][72];
  __shared__ _Float16 VT[64][72];     // transposed V chunk: VT[d][s]
  __shared__ _Float16 KS[64][72];     // K chunk, row-major (key, d)
  __shared__ _Float16 PCH[64][72];    // pos-emb chunk, row-major (pos, d)
  __shared__ _Float16 PS[4][16][72];  // per-wave exp(S) scratch (row-major)

  const int qc = blockIdx.x;  // 0..31 query chunk
  const int h  = blockIdx.y;  // head
  const int b  = blockIdx.z;  // batch
  const int tid  = threadIdx.x;
  const int w    = tid >> 5;
  const int lane = tid & 31;
  const int hh   = lane >> 4;
  const int ln   = lane & 15;
  const int t0   = qc * CKK;

  // Stage q + pos_bias_u / q + pos_bias_v for this chunk (64 x 64)
  {
    const int r  = tid >> 1;
    const int c0 = (tid & 1) * 32;
    const _Float16* src = Q + ((size_t)b * TT + t0 + r) * FF + h * DKK + c0;
#pragma unroll
    for (int i = 0; i < 32; ++i) {
      const int d = c0 + i;
      const float q = (float)src[i];
      QU[r][d] = (_Float16)(q + bias_u[h * DKK + d]);
      QV[r][d] = (_Float16)(q + bias_v[h * DKK + d]);
    }
  }
  __syncthreads();

  // Resident A-fragments for qu/qv (DK=64 -> 2 k-steps)
  const int arow = w * 16 + ln;
  v16h au0 = load_a_frag(&QU[arow][hh * 16],      &QU[arow][hh * 16 + 8]);
  v16h au1 = load_a_frag(&QU[arow][32 + hh * 16], &QU[arow][40 + hh * 16]);
  v16h av0 = load_a_frag(&QV[arow][hh * 16],      &QV[arow][hh * 16 + 8]);
  v16h av1 = load_a_frag(&QV[arow][32 + hh * 16], &QV[arow][40 + hh * 16]);

  v8f O0 = zero8(), O1 = zero8(), O2 = zero8(), O3 = zero8();
  float m8[8], l8[8];
#pragma unroll
  for (int r = 0; r < 8; ++r) { m8[r] = -1e30f; l8[r] = 0.f; }

  const int cs_lo = (qc >= NLL) ? (qc - NLL) : 0;
  for (int cs = cs_lo; cs <= qc; ++cs) {
    const int s0 = cs * CKK;
    const _Float16* Kb = Kh + ((size_t)b * TT + s0) * FF + h * DKK;
    const _Float16* Pb = Ph + (size_t)s0 * FF + h * DKK;

    __syncthreads();  // WAR: previous chunk done with staged buffers

    // Issue async DMA of K chunk + P chunk into LDS (512 16-half segments,
    // 4 per thread), then do the transposed V staging while the DMA runs.
#pragma unroll
    for (int it = 0; it < 4; ++it) {
      const int idx = it * 128 + tid;  // 0..511
      const int isP = idx >> 8;        // 0 = K, 1 = P
      const int r   = (idx >> 2) & 63;
      const int sg  = (idx & 3) * 16;
      const _Float16* src = (isP ? Pb : Kb) + (size_t)r * FF + sg;
      _Float16* dst = isP ? &PCH[r][sg] : &KS[r][sg];
      cp16_g2s(dst, src);
    }

    // Transposed V staging (VALU/DS work overlapping the async DMA)
    {
      const int s  = tid >> 1;
      const int c0 = (tid & 1) * 32;
      const _Float16* vsrc = Vh + ((size_t)b * TT + s0 + s) * FF + h * DKK + c0;
#pragma unroll
      for (int i = 0; i < 32; ++i) VT[c0 + i][s] = vsrc[i];
      // Prefetch next chunk's V rows into cache while we compute this one.
      if (cs < qc)
        __builtin_prefetch(Vh + ((size_t)b * TT + s0 + CKK + s) * FF + h * DKK + c0, 0, 1);
    }

    async_fence();
    __syncthreads();

    // Scores from LDS-resident K/P tiles: S[j] covers keys j*16..j*16+15
    v8f S[4];
#pragma unroll
    for (int j = 0; j < 4; ++j) {
      const int key = j * 16 + ln;
      v8f acc = zero8();
      acc = wmma16(au0, load_b_frag(&KS[key][hh * 16]), acc);
      acc = wmma16(av0, load_b_frag(&PCH[key][hh * 16]), acc);
      acc = wmma16(au1, load_b_frag(&KS[key][32 + hh * 16]), acc);
      acc = wmma16(av1, load_b_frag(&PCH[key][32 + hh * 16]), acc);
      S[j] = acc;
    }

    // Online softmax stats (row = hh*8 + r; reduce across the 16-lane half)
    float mnew[8], scv[8];
#pragma unroll
    for (int r = 0; r < 8; ++r) {
      float mx = -1e30f;
#pragma unroll
      for (int j = 0; j < 4; ++j) { S[j][r] *= 0.125f; mx = fmaxf(mx, S[j][r]); }
#pragma unroll
      for (int off = 8; off >= 1; off >>= 1) mx = fmaxf(mx, __shfl_xor(mx, off, 32));
      mnew[r] = fmaxf(m8[r], mx);
      scv[r]  = __expf(m8[r] - mnew[r]);
      m8[r]   = mnew[r];
    }
#pragma unroll
    for (int r = 0; r < 8; ++r) {
      float s = 0.f;
#pragma unroll
      for (int j = 0; j < 4; ++j) {
        const float e = __expf(S[j][r] - mnew[r]);
        S[j][r] = e;
        s += e;
      }
#pragma unroll
      for (int off = 8; off >= 1; off >>= 1) s += __shfl_xor(s, off, 32);
      l8[r] = l8[r] * scv[r] + s;
      O0[r] *= scv[r]; O1[r] *= scv[r]; O2[r] *= scv[r]; O3[r] *= scv[r];
    }

    // Spill P (C layout) to per-wave LDS row-major, reload as A fragments
#pragma unroll
    for (int j = 0; j < 4; ++j)
#pragma unroll
      for (int r = 0; r < 8; ++r)
        PS[w][hh * 8 + r][j * 16 + ln] = (_Float16)S[j][r];
    __syncthreads();

    // O += P @ V  (reduction over 64 keys = 2 k-steps)
#pragma unroll
    for (int ks = 0; ks < 2; ++ks) {
      v16h ap = load_a_frag(&PS[w][ln][ks * 32 + hh * 16],
                            &PS[w][ln][ks * 32 + hh * 16 + 8]);
      O0 = wmma16(ap, load_b_frag(&VT[0 * 16 + ln][ks * 32 + hh * 16]), O0);
      O1 = wmma16(ap, load_b_frag(&VT[1 * 16 + ln][ks * 32 + hh * 16]), O1);
      O2 = wmma16(ap, load_b_frag(&VT[2 * 16 + ln][ks * 32 + hh * 16]), O2);
      O3 = wmma16(ap, load_b_frag(&VT[3 * 16 + ln][ks * 32 + hh * 16]), O3);
    }
  }

  // Normalize and write X (B,T,F) f16
#pragma unroll
  for (int r = 0; r < 8; ++r) {
    const float inv = 1.f / l8[r];
    _Float16* xp = X + ((size_t)b * TT + t0 + w * 16 + hh * 8 + r) * FF + h * DKK;
    xp[0 * 16 + ln] = (_Float16)(O0[r] * inv);
    xp[1 * 16 + ln] = (_Float16)(O1[r] * inv);
    xp[2 * 16 + ln] = (_Float16)(O2[r] * inv);
    xp[3 * 16 + ln] = (_Float16)(O3[r] * inv);
  }
}

// ---------------------------------------------------------------------------
// Host launcher
// ---------------------------------------------------------------------------
extern "C" void kernel_launch(void* const* d_in, const int* in_sizes, int n_in,
                              void* d_out, int out_size, void* d_ws, size_t ws_size,
                              hipStream_t stream) {
  (void)in_sizes; (void)n_in; (void)out_size; (void)ws_size;

  const float* query   = (const float*)d_in[0];
  const float* key     = (const float*)d_in[1];
  const float* value   = (const float*)d_in[2];
  const float* pos_emb = (const float*)d_in[3];
  // d_in[4] = mask (implied by block-aligned chunk structure; unused)
  const float* Wq = (const float*)d_in[5];
  const float* bq = (const float*)d_in[6];
  const float* Wk = (const float*)d_in[7];
  const float* bk = (const float*)d_in[8];
  const float* Wv = (const float*)d_in[9];
  const float* bv = (const float*)d_in[10];
  const float* Wp = (const float*)d_in[11];
  const float* Wo = (const float*)d_in[12];
  const float* bo = (const float*)d_in[13];
  const float* bias_u = (const float*)d_in[14];
  const float* bias_v = (const float*)d_in[15];

  const size_t nBTF = (size_t)BB * TT * FF;  // 8,388,608
  _Float16* Qh = (_Float16*)d_ws;
  _Float16* Kh = Qh + nBTF;
  _Float16* Vh = Kh + nBTF;
  _Float16* Ph = Vh + nBTF;
  _Float16* Xh = Ph + (size_t)TT * FF;

  const dim3 gP(FF / 64, (BB * TT) / 64);  // (16,128)
  const dim3 gPos(FF / 64, TT / 64);       // (16,32)

  proj_gemm<float, true><<<gP, 256, 0, stream>>>(query, Wq, bq, Qh, BB * TT, FF, FF);
  proj_gemm<float, true><<<gP, 256, 0, stream>>>(key,   Wk, bk, Kh, BB * TT, FF, FF);
  proj_gemm<float, true><<<gP, 256, 0, stream>>>(value, Wv, bv, Vh, BB * TT, FF, FF);
  proj_gemm<float, true><<<gPos, 256, 0, stream>>>(pos_emb, Wp, nullptr, Ph, TT, FF, FF);

  attn_fused<<<dim3(TT / CKK, HH_, BB), 128, 0, stream>>>(Qh, Kh, Vh, Ph,
                                                          bias_u, bias_v, Xh);

  proj_gemm<_Float16, false><<<gP, 256, 0, stream>>>(Xh, Wo, bo, d_out, BB * TT, FF, FF);
}
